// MemoryEfficientAttention_81338090651920
// MI455X (gfx1250) — compile-verified, matrix operand order checked by reading, not run on hardware
//
#include <hip/hip_runtime.h>

// MI455X (gfx1250) flash-attention pipeline: f16 WMMA + f32 accumulate,
// TDM (tensor_load_to_lds) staging with hardware LDS padding, double-buffered K.
// ws layout: Q[h][4096][64] f16 (8MB) | K[h][4096][64] f16 (8MB) | att[4096][1024] f16 (8MB)

typedef __attribute__((ext_vector_type(16))) _Float16     v16h;
typedef __attribute__((ext_vector_type(8)))  _Float16     h8;
typedef __attribute__((ext_vector_type(8)))  float        v8f;
typedef __attribute__((ext_vector_type(4)))  unsigned int u32x4;
typedef __attribute__((ext_vector_type(8)))  unsigned int u32x8;

#define SEQ 4096
#define DM  1024
#define NH  16
#define DH  64

__device__ __forceinline__ v8f wmma16(v16h a, v16h b, v8f c) {
  // v_wmma_f32_16x16x32_f16: D = A(16x32 f16) x B(32x16 f16) + C(16x16 f32)
  return __builtin_amdgcn_wmma_f32_16x16x32_f16(false, a, false, b, (short)0, c, false, false);
}

// ---- TDM: DMA a (possibly 2D, row-padded) f16 tile from global into LDS ----
// tile_w elems per row, tile_h rows (0 => 1D), stride_elems between source rows.
// pi: pad interval code (2^(pi+1) DWORDs), pa: pad amount code (pa+1 DWORDs).
__device__ __forceinline__ void tdm_load_to_lds(const void* gsrc, unsigned lds_off,
                                                unsigned tile_w, unsigned tile_h,
                                                unsigned stride_elems,
                                                unsigned pi, unsigned pa) {
  unsigned long long ga = (unsigned long long)gsrc;
  unsigned td1 = tile_h ? tile_h : 1u;
  u32x4 g0;
  g0[0] = 1u;                                              // count=1, no gather
  g0[1] = lds_off;                                         // lds_addr
  g0[2] = (unsigned)ga;                                    // global_addr[31:0]
  g0[3] = (unsigned)((ga >> 32) & 0x1FFFFFFull) | 0x80000000u; // addr[56:32] | type=2
  u32x8 g1;
  g1[0] = (1u << 16) | (1u << 20) | (pi << 22) | (pa << 25); // data_size=2B, pad_enable
  g1[1] = (tile_w & 0xFFFFu) << 16;                        // tensor_dim0 lo16 (barrier addr 0)
  g1[2] = (tile_w >> 16) | ((td1 & 0xFFFFu) << 16);        // tensor_dim0 hi | tensor_dim1 lo
  g1[3] = (td1 >> 16) | ((tile_w & 0xFFFFu) << 16);        // tensor_dim1 hi | tile_dim0
  g1[4] = tile_h;                                          // tile_dim1 (0=1D), tile_dim2=0
  g1[5] = stride_elems;                                    // tensor_dim0_stride[31:0]
  g1[6] = 0u;
  g1[7] = 0u;
  asm volatile("tensor_load_to_lds %0, %1" :: "s"(g0), "s"(g1) : "memory");
}

// A-style fragment (also the transposed-B read): lane&15 -> row index,
// element e -> k = (e&8?16:0) + half*8 + (e&7). Source is row-major [i][k], stride ld.
__device__ __forceinline__ v16h frag_rowmajor(const _Float16* s, int ld, int i0, int k0, int lane) {
  const int hf = (lane >> 4) & 1;
  const _Float16* row = s + (size_t)(i0 + (lane & 15)) * ld + k0 + hf * 8;
  v16h f;
#pragma unroll
  for (int e = 0; e < 16; e += 2) {
    int k = ((e & 8) ? 16 : 0) + (e & 7);
    f[e]     = row[k];
    f[e + 1] = row[k + 1];
  }
  return f;
}

// B fragment from natural row-major [k][n] storage: lane&15 -> n.
__device__ __forceinline__ v16h frag_colmajor(const _Float16* s, int ld, int k0, int n0, int lane) {
  const int hf = (lane >> 4) & 1;
  const _Float16* col = s + (size_t)(k0 + hf * 8) * ld + n0 + (lane & 15);
  v16h f;
#pragma unroll
  for (int e = 0; e < 16; ++e) {
    int k = ((e & 8) ? 16 : 0) + (e & 7);
    f[e] = col[(size_t)k * ld];
  }
  return f;
}

// ---------------- Kernel 1: fused QK projection (skips unused V) ----------------
__global__ __launch_bounds__(256) void qkv_proj(const float* __restrict__ x,
                                                const float* __restrict__ w,
                                                _Float16* __restrict__ Qw,
                                                _Float16* __restrict__ Kw) {
  __shared__ alignas(16) _Float16 As[128 * 40];   // 128 rows x 32 k, padded
  __shared__ alignas(16) _Float16 Bs[32 * 136];   // 32 k x 128 n, padded
  const int t = threadIdx.x, lane = t & 31, wave = t >> 5;
  const int wm = wave >> 2, wn = wave & 3;              // 2x4 wave grid, 64x32 per wave
  const int mb = blockIdx.x * 128, nb = blockIdx.y * 128;
  v8f acc[4][2];
#pragma unroll
  for (int i = 0; i < 4; ++i)
#pragma unroll
    for (int j = 0; j < 2; ++j)
#pragma unroll
      for (int r = 0; r < 8; ++r) acc[i][j][r] = 0.f;

  for (int k0 = 0; k0 < DM; k0 += 32) {
    __syncthreads();
    // Issue all 8 staged loads first (independent register quads -> loads stay in
    // flight together), then convert+store: hides VMEM latency behind VALU cvt.
    float4 va[4], vb[4];
#pragma unroll
    for (int i = 0; i < 4; ++i) {               // x tile: 128x32 f32
      int flat = t + i * 256;
      int row = flat >> 3, c = (flat & 7) * 4;
      va[i] = *(const float4*)(x + (size_t)(mb + row) * DM + k0 + c);
    }
#pragma unroll
    for (int i = 0; i < 4; ++i) {               // w_qkv tile: 32x128 f32
      int flat = t + i * 256;
      int kr = flat >> 5, c = (flat & 31) * 4;
      vb[i] = *(const float4*)(w + (size_t)(k0 + kr) * 3072 + nb + c);
    }
    if (k0 + 32 < DM)                           // hint next weight panel toward L2
      __builtin_prefetch(w + (size_t)(k0 + 32) * 3072 + nb + (t & 31) * 4, 0, 1);
#pragma unroll
    for (int i = 0; i < 4; ++i) {
      int flat = t + i * 256;
      int row = flat >> 3, c = (flat & 7) * 4;
      _Float16* d = &As[row * 40 + c];
      d[0] = (_Float16)va[i].x; d[1] = (_Float16)va[i].y;
      d[2] = (_Float16)va[i].z; d[3] = (_Float16)va[i].w;
    }
#pragma unroll
    for (int i = 0; i < 4; ++i) {
      int flat = t + i * 256;
      int kr = flat >> 5, c = (flat & 31) * 4;
      _Float16* d = &Bs[kr * 136 + c];
      d[0] = (_Float16)vb[i].x; d[1] = (_Float16)vb[i].y;
      d[2] = (_Float16)vb[i].z; d[3] = (_Float16)vb[i].w;
    }
    __syncthreads();
    v16h af[4], bf[2];
#pragma unroll
    for (int mt = 0; mt < 4; ++mt) af[mt] = frag_rowmajor(As, 40, wm * 64 + mt * 16, 0, lane);
#pragma unroll
    for (int nt = 0; nt < 2; ++nt) bf[nt] = frag_colmajor(Bs, 136, 0, wn * 32 + nt * 16, lane);
#pragma unroll
    for (int mt = 0; mt < 4; ++mt)
#pragma unroll
      for (int nt = 0; nt < 2; ++nt) acc[mt][nt] = wmma16(af[mt], bf[nt], acc[mt][nt]);
  }

  const int ln = lane & 15, hf = (lane >> 4) & 1;
#pragma unroll
  for (int mt = 0; mt < 4; ++mt)
#pragma unroll
    for (int nt = 0; nt < 2; ++nt)
#pragma unroll
      for (int r = 0; r < 8; ++r) {
        int grow = mb + wm * 64 + mt * 16 + hf * 8 + r;
        int gcol = nb + wn * 32 + nt * 16 + ln;
        float v = acc[mt][nt][r];
        if (gcol < DM) {            // Q, pre-scaled by d^-0.5 = 0.125
          Qw[(size_t)(gcol >> 6) * SEQ * DH + (size_t)grow * DH + (gcol & 63)] = (_Float16)(v * 0.125f);
        } else {                    // K
          int c = gcol - DM;
          Kw[(size_t)(c >> 6) * SEQ * DH + (size_t)grow * DH + (c & 63)] = (_Float16)v;
        }
      }
}

// ---------------- Kernel 2: flash attention (V := K, per the reference) ----------------
__global__ __launch_bounds__(128) void flash_attn(const _Float16* __restrict__ Qw,
                                                  const _Float16* __restrict__ Kw,
                                                  _Float16* __restrict__ att) {
  __shared__ alignas(16) _Float16 PQ[128 * 72];     // Q staging, then per-wave P tiles
  __shared__ alignas(16) _Float16 Ks[2][64 * 72];   // double-buffered K blocks
  const int t = threadIdx.x, lane = t & 31, wave = t >> 5;   // 4 waves x 32 rows
  const int qb = blockIdx.x * 128;
  const int head = blockIdx.y;
  const _Float16* Qh = Qw + (size_t)head * SEQ * DH;
  const _Float16* Kh = Kw + (size_t)head * SEQ * DH;

  const unsigned pqOff  = (unsigned)(unsigned long long)(void*)&PQ[0];
  const unsigned ksOff0 = (unsigned)(unsigned long long)(void*)&Ks[0][0];
  const unsigned ksOff1 = (unsigned)(unsigned long long)(void*)&Ks[1][0];

  // TDM: stage Q tile (128x64, contiguous) and K block 0, both with HW row padding
  if (t < 32) {
    tdm_load_to_lds(Qh + (size_t)qb * DH, pqOff, 128 * DH, 0, 128 * DH, 4, 3);
    tdm_load_to_lds(Kh, ksOff0, 64 * DH, 0, 64 * DH, 4, 3);
    __builtin_amdgcn_s_wait_tensorcnt(0);
  }
  __syncthreads();

  v16h qf[2][2];                                 // Q fragments held for whole kernel
#pragma unroll
  for (int mt = 0; mt < 2; ++mt)
#pragma unroll
    for (int ks = 0; ks < 2; ++ks)
      qf[mt][ks] = frag_rowmajor(PQ, 72, wave * 32 + mt * 16, ks * 32, lane);
  __syncthreads();

  v8f O[2][4];
  float mrun[2][8], lrun[2][8];
#pragma unroll
  for (int mt = 0; mt < 2; ++mt)
#pragma unroll
    for (int nt = 0; nt < 4; ++nt)
#pragma unroll
      for (int r = 0; r < 8; ++r) O[mt][nt][r] = 0.f;
#pragma unroll
  for (int mt = 0; mt < 2; ++mt)
#pragma unroll
    for (int r = 0; r < 8; ++r) { mrun[mt][r] = -3.0e38f; lrun[mt][r] = 0.f; }

  const int ln = lane & 15, hf = (lane >> 4) & 1;

  for (int kb = 0; kb < SEQ / 64; ++kb) {
    // async-prefetch next K block into the other buffer while we compute
    if (t < 32 && kb + 1 < SEQ / 64)
      tdm_load_to_lds(Kh + (size_t)(kb + 1) * 64 * DH,
                      ((kb + 1) & 1) ? ksOff1 : ksOff0, 64 * DH, 0, 64 * DH, 4, 3);
    const _Float16* Kc = &Ks[kb & 1][0];

    // S = Q * K^T : B[k][n] = K[n][k] -> transposed read via frag_rowmajor
    v8f S[2][4];
#pragma unroll
    for (int mt = 0; mt < 2; ++mt)
#pragma unroll
      for (int nt = 0; nt < 4; ++nt)
#pragma unroll
        for (int r = 0; r < 8; ++r) S[mt][nt][r] = 0.f;
#pragma unroll
    for (int ks = 0; ks < 2; ++ks) {
      v16h bf[4];
#pragma unroll
      for (int nt = 0; nt < 4; ++nt) bf[nt] = frag_rowmajor(Kc, 72, nt * 16, ks * 32, lane);
#pragma unroll
      for (int mt = 0; mt < 2; ++mt)
#pragma unroll
        for (int nt = 0; nt < 4; ++nt) S[mt][nt] = wmma16(qf[mt][ks], bf[nt], S[mt][nt]);
    }

    // online softmax: each C/D row m=r+hf*8 lives in one 16-lane half
#pragma unroll
    for (int mt = 0; mt < 2; ++mt)
#pragma unroll
      for (int r = 0; r < 8; ++r) {
        float mx = -3.0e38f;
#pragma unroll
        for (int nt = 0; nt < 4; ++nt) mx = fmaxf(mx, S[mt][nt][r]);
#pragma unroll
        for (int off = 1; off < 16; off <<= 1) mx = fmaxf(mx, __shfl_xor(mx, off, 32));
        float mnew = fmaxf(mrun[mt][r], mx);
        float alpha = __expf(mrun[mt][r] - mnew);
        float sum = 0.f;
#pragma unroll
        for (int nt = 0; nt < 4; ++nt) {
          float p = __expf(S[mt][nt][r] - mnew);
          S[mt][nt][r] = p;
          sum += p;
        }
#pragma unroll
        for (int off = 1; off < 16; off <<= 1) sum += __shfl_xor(sum, off, 32);
        lrun[mt][r] = lrun[mt][r] * alpha + sum;
        mrun[mt][r] = mnew;
#pragma unroll
        for (int nt = 0; nt < 4; ++nt) O[mt][nt][r] *= alpha;
      }

    // D-layout -> A-layout conversion through this wave's private LDS rows
#pragma unroll
    for (int mt = 0; mt < 2; ++mt)
#pragma unroll
      for (int nt = 0; nt < 4; ++nt)
#pragma unroll
        for (int r = 0; r < 8; ++r)
          PQ[(wave * 32 + mt * 16 + hf * 8 + r) * 72 + nt * 16 + ln] = (_Float16)S[mt][nt][r];
    asm volatile("s_wait_dscnt 0x0" ::: "memory");   // same-wave LDS RAW

    // O += P * K   (value matrix is K, faithful to reference)
#pragma unroll
    for (int ks = 0; ks < 2; ++ks) {
      v16h af[2], bf[4];
#pragma unroll
      for (int mt = 0; mt < 2; ++mt) af[mt] = frag_rowmajor(PQ, 72, wave * 32 + mt * 16, ks * 32, lane);
#pragma unroll
      for (int nt = 0; nt < 4; ++nt) bf[nt] = frag_colmajor(Kc, 72, ks * 32, nt * 16, lane);
#pragma unroll
      for (int mt = 0; mt < 2; ++mt)
#pragma unroll
        for (int nt = 0; nt < 4; ++nt) O[mt][nt] = wmma16(af[mt], bf[nt], O[mt][nt]);
    }

    if (t < 32) __builtin_amdgcn_s_wait_tensorcnt(0);  // next block landed
    __syncthreads();                                   // everyone done with Kc + buffer swap
  }

#pragma unroll
  for (int mt = 0; mt < 2; ++mt)
#pragma unroll
    for (int nt = 0; nt < 4; ++nt)
#pragma unroll
      for (int r = 0; r < 8; ++r) {
        int grow = qb + wave * 32 + mt * 16 + hf * 8 + r;
        float v = O[mt][nt][r] / lrun[mt][r];
        att[(size_t)grow * DM + head * DH + nt * 16 + ln] = (_Float16)v;
      }
}

// ---------------- Kernel 3: output projection + bias ----------------
__global__ __launch_bounds__(256) void out_proj(const _Float16* __restrict__ att,
                                                const float* __restrict__ w,
                                                const float* __restrict__ bias,
                                                float* __restrict__ out) {
  __shared__ alignas(16) _Float16 As[128 * 40];
  __shared__ alignas(16) _Float16 Bs[32 * 136];
  const int t = threadIdx.x, lane = t & 31, wave = t >> 5;
  const int wm = wave >> 2, wn = wave & 3;
  const int mb = blockIdx.x * 128, nb = blockIdx.y * 128;
  const unsigned asOff = (unsigned)(unsigned long long)(void*)&As[0];
  v8f acc[4][2];
#pragma unroll
  for (int i = 0; i < 4; ++i)
#pragma unroll
    for (int j = 0; j < 2; ++j)
#pragma unroll
      for (int r = 0; r < 8; ++r) acc[i][j][r] = 0.f;

  for (int k0 = 0; k0 < DM; k0 += 32) {
    __syncthreads();
    // TDM: 2D f16 tile 32x128 from att (stride DM), HW-padded to 40-half rows,
    // overlapped with the f32->f16 B-panel conversion below.
    if (t < 32)
      tdm_load_to_lds(att + (size_t)mb * DM + k0, asOff, 32, 128, DM, 3, 3);
    float4 vb[4];
#pragma unroll
    for (int i = 0; i < 4; ++i) {               // w_out tile: 32x128 f32, loads batched
      int flat = t + i * 256;
      int kr = flat >> 5, c = (flat & 31) * 4;
      vb[i] = *(const float4*)(w + (size_t)(k0 + kr) * DM + nb + c);
    }
#pragma unroll
    for (int i = 0; i < 4; ++i) {
      int flat = t + i * 256;
      int kr = flat >> 5, c = (flat & 31) * 4;
      _Float16* d = &Bs[kr * 136 + c];
      d[0] = (_Float16)vb[i].x; d[1] = (_Float16)vb[i].y;
      d[2] = (_Float16)vb[i].z; d[3] = (_Float16)vb[i].w;
    }
    if (t < 32) __builtin_amdgcn_s_wait_tensorcnt(0);
    __syncthreads();
    v16h af[4], bf[2];
#pragma unroll
    for (int mt = 0; mt < 4; ++mt) af[mt] = frag_rowmajor(As, 40, wm * 64 + mt * 16, 0, lane);
#pragma unroll
    for (int nt = 0; nt < 2; ++nt) bf[nt] = frag_colmajor(Bs, 136, 0, wn * 32 + nt * 16, lane);
#pragma unroll
    for (int mt = 0; mt < 4; ++mt)
#pragma unroll
      for (int nt = 0; nt < 2; ++nt) acc[mt][nt] = wmma16(af[mt], bf[nt], acc[mt][nt]);
  }

  const int ln = lane & 15, hf = (lane >> 4) & 1;
#pragma unroll
  for (int mt = 0; mt < 4; ++mt)
#pragma unroll
    for (int nt = 0; nt < 2; ++nt)
#pragma unroll
      for (int r = 0; r < 8; ++r) {
        int grow = mb + wm * 64 + mt * 16 + hf * 8 + r;
        int gcol = nb + wn * 32 + nt * 16 + ln;
        out[(size_t)grow * DM + gcol] = acc[mt][nt][r] + bias[gcol];
      }
}

extern "C" void kernel_launch(void* const* d_in, const int* in_sizes, int n_in,
                              void* d_out, int out_size, void* d_ws, size_t ws_size,
                              hipStream_t stream) {
  (void)in_sizes; (void)n_in; (void)out_size; (void)ws_size;
  const float* x     = (const float*)d_in[0];
  const float* w_qkv = (const float*)d_in[1];
  const float* w_out = (const float*)d_in[2];
  const float* b_out = (const float*)d_in[3];
  float* out = (float*)d_out;

  _Float16* Qw  = (_Float16*)d_ws;                       // 16*4096*64 f16
  _Float16* Kw  = Qw + (size_t)NH * SEQ * DH;            // 16*4096*64 f16
  _Float16* att = Kw + (size_t)NH * SEQ * DH;            // 8MB f16

  qkv_proj<<<dim3(SEQ / 128, (2 * DM) / 128), 256, 0, stream>>>(x, w_qkv, Qw, Kw);
  flash_attn<<<dim3(SEQ / 128, NH), 128, 0, stream>>>(Qw, Kw, att);
  out_proj<<<dim3(SEQ / 128, DM / 128), 256, 0, stream>>>(att, w_out, b_out, out);
}
